// SOC_1726576855909
// MI455X (gfx1250) — matrix operand order repaired and checked by reading, not compiled, not run on hardware
//
#include <hip/hip_runtime.h>

// ---------------------------------------------------------------------------
// Episode constants (match reference shape_source)
// ---------------------------------------------------------------------------
#define NB      8       // batch_size (episodes)
#define NWAY    5
#define NQRY    75
#define NP      9       // patches per sample
#define CCH     640     // backbone channels
#define KDIM    192     // 3*8*8 conv reduction
#define NPATCH  16      // 4x4 conv output positions
#define N_IMG   (NB * (NWAY + NQRY) * NP)   // 5760 conv problems
#define MROWS   48      // 45 support rows padded to 48
#define NROWS   688     // 675 query rows padded to 688
#define BT_ST   208     // LDS patch-tile stride (halves), padded vs 192
#define CV_ST   17      // LDS conv-out stride (dwords), padded vs 16

typedef __attribute__((ext_vector_type(16))) __bf16        v16bf;
typedef __attribute__((ext_vector_type(8)))  float         v8f;
typedef __attribute__((ext_vector_type(4)))  unsigned int  u32x4;

union Frag { v16bf v; u32x4 q[2]; };

__device__ __forceinline__ unsigned short f2bf(float f) {
  unsigned int u = __float_as_uint(f);
  u += 0x7FFFu + ((u >> 16) & 1u);          // round-to-nearest-even
  return (unsigned short)(u >> 16);
}

// ---------------------------------------------------------------------------
// Kernel 0: conv weights f32 -> bf16 (OIHW flat == [640][192] im2col order)
// ---------------------------------------------------------------------------
__global__ void k_w2bf(const float* __restrict__ w, unsigned short* __restrict__ wbf) {
  int i = blockIdx.x * 256 + threadIdx.x;
  if (i < CCH * KDIM) wbf[i] = f2bf(w[i]);
}

// ---------------------------------------------------------------------------
// Kernel 1: per image n (5760 WGs, 8 waves):
//   async-copy raw image (12 KB f32) HBM -> LDS on the ASYNCcnt path,
//   repack to bf16 patch tile, then
//   WMMA GEMM  W[640x192] x patches[192x16] -> conv[640x16] (LDS, f32)
//   -> per-column L2 norm over 640 ch -> mean over 16 cols -> row L2 norm
//   -> bf16 row written into ftrb / fteb for the similarity GEMM.
//   sConv doubles as the raw-image staging buffer (disjoint lifetimes:
//   raw reads finish before barrier 1; sConv writes start after it).
// ---------------------------------------------------------------------------
__global__ __launch_bounds__(256) void k_conv_feat(
    const float* __restrict__ x,              // [5760][3][32][32]
    const unsigned short* __restrict__ wbf,   // [640][192] bf16 (L2-resident)
    unsigned short* __restrict__ ftrb,        // [8][48][640]  bf16
    unsigned short* __restrict__ fteb)        // [8][688][640] bf16
{
  __shared__ __align__(16) unsigned short sBt[NPATCH * BT_ST];  // patch tile, bf16
  __shared__ __align__(16) float sConv[CCH * CV_ST];            // conv out / raw stage
  __shared__ float sSsq[16];
  __shared__ float sInv[16];
  __shared__ float sRss;

  const int t = threadIdx.x;
  const int n = blockIdx.x;
  const float* img = x + (size_t)n * 3072;

  if (t < 16) sSsq[t] = 0.f;
  if (t == 0) sRss = 0.f;

  // --- async HBM -> LDS copy of the raw image (768 x 16B), ASYNCcnt path ---
  for (int q = t; q < 768; q += 256) {
    const int e = q * 4;                                   // f32 index, 16B grain
    unsigned ldsoff = (unsigned)(size_t)(&sConv[e]);       // addr[31:0] == LDS offset
    unsigned long long ga = (unsigned long long)(size_t)(img + e);
    asm volatile("global_load_async_to_lds_b128 %0, %1, off"
                 :: "v"(ldsoff), "v"(ga) : "memory");
  }
  asm volatile("s_wait_asynccnt 0x0" ::: "memory");        // own-wave copies done
  __syncthreads();                                         // visible to all waves

  // repack raw f32 image -> bf16 B-tile sBt[patch][k]
  for (int e = t; e < 3072; e += 256) {
    float v  = sConv[e];
    int i    = e >> 10;            // input channel
    int y    = (e >> 5) & 31;
    int xc   = e & 31;
    int pidx = (y >> 3) * 4 + (xc >> 3);
    int k    = i * 64 + (y & 7) * 8 + (xc & 7);
    sBt[pidx * BT_ST + k] = f2bf(v);
  }
  __syncthreads();                                         // barrier 1

  const int wv   = t >> 5;
  const int lane = t & 31;
  const int colh = lane & 15;
  const bool hi  = lane >= 16;

  v8f acc[5] = {};
  const unsigned short* btcol = sBt + colh * BT_ST;   // B: lane holds column pidx

  #pragma unroll
  for (int ks = 0; ks < 6; ++ks) {                    // K = 192 = 6 x 32
    Frag bfr;
    const int kb = ks * 32 + (hi ? 8 : 0);            // K + 0..7,16..23 (or 8..15,24..31)
    bfr.q[0] = *(const u32x4*)(btcol + kb);
    bfr.q[1] = *(const u32x4*)(btcol + kb + 16);
    #pragma unroll
    for (int j = 0; j < 5; ++j) {                     // 5 M-tiles per wave
      const int mrow = (wv * 5 + j) * 16 + colh;
      const unsigned short* ap = wbf + mrow * KDIM + ks * 32 + (hi ? 8 : 0);
      Frag afr;
      afr.q[0] = *(const u32x4*)(ap);                 // K + 0..7  (or 8..15)
      afr.q[1] = *(const u32x4*)(ap + 16);            // K + 16..23 (or 24..31)
      acc[j] = __builtin_amdgcn_wmma_f32_16x16x32_bf16(
          false, afr.v, false, bfr.v, (short)0, acc[j], false, false);
    }
  }

  // C fragments -> LDS conv[channel][patch]
  #pragma unroll
  for (int j = 0; j < 5; ++j) {
    const int mbase = (wv * 5 + j) * 16 + (hi ? 8 : 0);
    #pragma unroll
    for (int r = 0; r < 8; ++r)
      sConv[(mbase + r) * CV_ST + colh] = acc[j][r];
  }
  __syncthreads();

  // per-patch (column) L2 norm over 640 channels
  {
    const int col = t & 15, rg = t >> 4;
    float p = 0.f;
    for (int row = rg; row < CCH; row += 16) {
      float v = sConv[row * CV_ST + col];
      p += v * v;
    }
    atomicAdd(&sSsq[col], p);
  }
  __syncthreads();
  if (t < 16) sInv[t] = 1.0f / fmaxf(sqrtf(sSsq[t]), 1e-12f);
  __syncthreads();

  // normalize + mean-pool 16 patches; accumulate row sum-of-squares
  float pv[3];
  int   pc = 0;
  float rpart = 0.f;
  for (int c = t; c < CCH; c += 256) {
    float s = 0.f;
    #pragma unroll
    for (int j = 0; j < 16; ++j) s += sConv[c * CV_ST + j] * sInv[j];
    s *= (1.0f / 16.0f);
    pv[pc++] = s;
    rpart += s * s;
  }
  atomicAdd(&sRss, rpart);
  __syncthreads();
  const float rscale = 1.0f / fmaxf(sqrtf(sRss), 1e-12f);

  // scatter normalized bf16 row into support / query GEMM operand buffers
  const int imgIdx = n / NP, pp = n % NP;
  const int b = imgIdx / (NWAY + NQRY), s = imgIdx % (NWAY + NQRY);
  unsigned short* dst = (s < NWAY)
      ? ftrb + ((size_t)(b * MROWS + s * NP + pp)) * CCH
      : fteb + ((size_t)(b * NROWS + (s - NWAY) * NP + pp)) * CCH;
  pc = 0;
  for (int c = t; c < CCH; c += 256) dst[c] = f2bf(pv[pc++] * rscale);
}

// ---------------------------------------------------------------------------
// Kernel 2: sim_b = Ftr_b[48x640] . Fte_b[688x640]^T  (one wave per 16x16 tile)
//   8 batches x 3 x 43 tiles = 1032 waves = 129 blocks of 8 waves, K = 20x32.
// ---------------------------------------------------------------------------
__global__ __launch_bounds__(256) void k_sim(
    const unsigned short* __restrict__ ftrb,
    const unsigned short* __restrict__ fteb,
    float* __restrict__ simb)                 // [8][48][688] f32
{
  const int t = threadIdx.x;
  const int wv = t >> 5, lane = t & 31;
  const int g = blockIdx.x * 8 + wv;          // 0..1031
  const int b = g / 129;
  const int r = g % 129;
  const int mt = r / 43, nt = r % 43;
  const int colh = lane & 15;
  const bool hi  = lane >= 16;

  const unsigned short* arow = ftrb + ((size_t)(b * MROWS + mt * 16 + colh)) * CCH;
  const unsigned short* brow = fteb + ((size_t)(b * NROWS + nt * 16 + colh)) * CCH;

  v8f acc = {};
  for (int ks = 0; ks < 20; ++ks) {
    Frag af, bf;
    const unsigned short* ap = arow + ks * 32 + (hi ? 8 : 0);
    af.q[0] = *(const u32x4*)(ap);
    af.q[1] = *(const u32x4*)(ap + 16);
    const unsigned short* bp = brow + ks * 32 + (hi ? 8 : 0);
    bf.q[0] = *(const u32x4*)(bp);
    bf.q[1] = *(const u32x4*)(bp + 16);
    acc = __builtin_amdgcn_wmma_f32_16x16x32_bf16(
        false, af.v, false, bf.v, (short)0, acc, false, false);
  }

  const int rowb = mt * 16 + (hi ? 8 : 0);
  const int col  = nt * 16 + colh;
  float* out = simb + (size_t)b * MROWS * NROWS;
  #pragma unroll
  for (int rr = 0; rr < 8; ++rr)
    out[(rowb + rr) * NROWS + col] = acc[rr];
}

// ---------------------------------------------------------------------------
// Kernel 3: greedy 9-step row/col pruning per (b, query, support) = 3000 items
// ---------------------------------------------------------------------------
__global__ void k_greedy(const float* __restrict__ simb, float* __restrict__ out) {
  const int tid = blockIdx.x * blockDim.x + threadIdx.x;
  if (tid >= NB * NQRY * NWAY) return;
  const int b   = tid / (NQRY * NWAY);
  const int rem = tid % (NQRY * NWAY);
  const int nq  = rem / NWAY;
  const int m   = rem % NWAY;

  float s[81];
  const float* base = simb + ((size_t)(b * MROWS + m * NP)) * NROWS + nq * NP;
  #pragma unroll
  for (int h = 0; h < 9; ++h)
    #pragma unroll
    for (int w = 0; w < 9; ++w)
      s[h * 9 + w] = base[h * NROWS + w];

  unsigned rm = 0x1FFu, cm = 0x1FFu;
  float total = 0.f, bp = 1.0f;
  for (int it = 0; it < 9; ++it) {
    float best = -1e30f;
    int br = 0, bc = 0;
    #pragma unroll
    for (int h = 0; h < 9; ++h)
      if (rm & (1u << h))
        #pragma unroll
        for (int w = 0; w < 9; ++w)
          if (cm & (1u << w)) {
            const float v = s[h * 9 + w];
            if (v > best) { best = v; br = h; bc = w; }  // first-max == argmax
          }
    total += fmaxf(best, 0.f) * bp;   // relu(max) * beta^it
    bp *= 0.5f;                       // BETA = 0.5
    rm &= ~(1u << br);
    cm &= ~(1u << bc);
  }
  out[tid] = total;                   // [b][nq][m] flat
}

// ---------------------------------------------------------------------------
// Launch: ws layout  wbf(240KB) | ftrb(480KB) | fteb(6.72MB) | sim(1.03MB)
// ---------------------------------------------------------------------------
extern "C" void kernel_launch(void* const* d_in, const int* in_sizes, int n_in,
                              void* d_out, int out_size, void* d_ws, size_t ws_size,
                              hipStream_t stream) {
  (void)in_sizes; (void)n_in; (void)out_size; (void)ws_size;
  const float* data   = (const float*)d_in[0];
  const float* conv_w = (const float*)d_in[1];
  float* out = (float*)d_out;
  char*  ws  = (char*)d_ws;

  unsigned short* wbf  = (unsigned short*)(ws + 0);        // 640*192*2   = 245760
  unsigned short* ftrb = (unsigned short*)(ws + 245760);   // 8*48*640*2  = 491520
  unsigned short* fteb = (unsigned short*)(ws + 737280);   // 8*688*640*2 = 7045120
  float*          simb = (float*)(ws + 7782400);           // 8*48*688*4  = 1056768

  k_w2bf    <<<(CCH * KDIM + 255) / 256, 256, 0, stream>>>(conv_w, wbf);
  k_conv_feat<<<N_IMG, 256, 0, stream>>>(data, wbf, ftrb, fteb);
  k_sim     <<<129, 256, 0, stream>>>(ftrb, fteb, simb);
  k_greedy  <<<(NB * NQRY * NWAY + 255) / 256, 256, 0, stream>>>(simb, out);
}
